// Attention_59983513256168
// MI455X (gfx1250) — matrix operand order repaired and checked by reading, not compiled, not run on hardware
//
#include <hip/hip_runtime.h>

// ---------------------------------------------------------------------------
// Multi-head attention forward for MI455X (gfx1250, wave32, WMMA + TDM).
//   x:      [4, 2048, 1024] f32
//   W_qkv:  [1024, 3072]    f32
//   W_proj: [1024, 1024]    f32
//   b_proj: [1024]          f32
//   out:    [4, 2048, 1024] f32
// Pipeline: f32->f16 convert -> QKV GEMM (wmma f16/f32 acc, TDM A-tile
// staging) -> flash attention (wmma + online softmax, TDM K-tile staging)
// -> proj GEMM + bias.
// ---------------------------------------------------------------------------

typedef __attribute__((ext_vector_type(16))) _Float16 v16h;
typedef __attribute__((ext_vector_type(8)))  _Float16 v8h;
typedef __attribute__((ext_vector_type(4)))  _Float16 v4h;
typedef __attribute__((ext_vector_type(8)))  float    v8f;

#define DIMC   1024
#define NHEAD  16
#define HDIM   64
#define BATCH  4
#define SEQN   2048
#define QKSCALE 0.125f   // 64^-0.5

// ---- Tensor Data Mover (gfx1250) ------------------------------------------
#if defined(__HIP_DEVICE_COMPILE__) && defined(__gfx1250__) && \
    __has_builtin(__builtin_amdgcn_tensor_load_to_lds) &&      \
    __has_builtin(__builtin_amdgcn_s_wait_tensorcnt)
#define HAS_TDM 1
#else
#define HAS_TDM 0
#endif

#if HAS_TDM
typedef __attribute__((ext_vector_type(4))) unsigned tdm_v4u;
typedef __attribute__((ext_vector_type(8))) int      tdm_v8i;
typedef __attribute__((ext_vector_type(4))) int      tdm_v4i;

// Low 32 bits of a generic LDS pointer == byte offset in the wave's LDS.
__device__ __forceinline__ unsigned lds_off_u32(const void* p) {
  return (unsigned)(unsigned long long)p;
}

// 2-D tile DMA: tile_h rows of tile_w f16 elements, global row stride
// row_stride_h elements, LDS destination padded per pad codes
// (pad_interval: DWORDs-between-pads = 2<<code; pad_amount: DWORDs = code+1).
__device__ __forceinline__ void tdm_load_2d(unsigned lds_off, const void* gsrc,
                                            unsigned tile_w, unsigned tile_h,
                                            unsigned row_stride_h,
                                            unsigned pad_int_code,
                                            unsigned pad_amt_code) {
  unsigned long long ga = (unsigned long long)gsrc;
  tdm_v4u g0;
  g0.x = 1u;                                  // count=1, user descriptor
  g0.y = lds_off;                             // lds_addr (bytes)
  g0.z = (unsigned)ga;                        // global_addr[31:0]
  g0.w = (unsigned)((ga >> 32) & 0x01FFFFFFull) | (2u << 30);  // addr[56:32]|type=2
  const unsigned dim0 = 1u << 20;             // huge tensor dims: no OOB clipping
  const unsigned dim1 = 1u << 20;
  tdm_v8i g1;
  g1[0] = (int)((1u << 16) |                  // data_size = 2 bytes
                (1u << 20) |                  // pad_enable
                (pad_int_code << 22) | (pad_amt_code << 25));
  g1[1] = (int)((dim0 & 0xffffu) << 16);                       // tensor_dim0 lo16
  g1[2] = (int)((dim0 >> 16) | ((dim1 & 0xffffu) << 16));      // dim0 hi / dim1 lo
  g1[3] = (int)((dim1 >> 16) | (tile_w << 16));                // dim1 hi / tile_dim0
  g1[4] = (int)(tile_h & 0xffffu);                             // tile_dim1
  g1[5] = (int)row_stride_h;                                   // dim0_stride lo32
  g1[6] = 0;
  g1[7] = 0;
  tdm_v4i z4 = {};
  tdm_v8i z8 = {};
  // clang-23 / therock-10.0 form: (g0, g1, g2, g3, extra, cpol)
  __builtin_amdgcn_tensor_load_to_lds(g0, g1, z4, z4, z8, 0);
}
#endif

// ---- WMMA fragment helpers (CDNA5 16x16x32 f16, wave32) -------------------
// A (16x32, MxK): lane m = lane&15; kb = (lane>=16)?8:0;
//   halves[0..7] = A[m][kb+0..7], halves[8..15] = A[m][kb+16..kb+23]
__device__ __forceinline__ v16h ld_a_frag(const _Float16* p) {
  v8h lo = *(const v8h*)(p);
  v8h hi = *(const v8h*)(p + 16);
  return __builtin_shufflevector(lo, hi, 0,1,2,3,4,5,6,7,8,9,10,11,12,13,14,15);
}
// B (32x16, KxN) stored N-major (Bt[n][k]): lane n = lane&15; kb=(lane>=16)?8:0;
//   halves[0..7] = Bt[n][kb+0..7], halves[8..15] = Bt[n][kb+16..kb+23]
__device__ __forceinline__ v16h ld_b_frag(const _Float16* p) {
  v8h lo = *(const v8h*)(p);
  v8h hi = *(const v8h*)(p + 16);
  return __builtin_shufflevector(lo, hi, 0,1,2,3,4,5,6,7,8,9,10,11,12,13,14,15);
}
__device__ __forceinline__ v8f mma16(v16h a, v16h b, v8f c) {
  return __builtin_amdgcn_wmma_f32_16x16x32_f16(false, a, false, b, (short)0, c,
                                                false, false);
}

// ---- f32 -> f16 conversion (vectorized x4) --------------------------------
__global__ __launch_bounds__(256) void f32_to_f16_kernel(
    const float* __restrict__ src, _Float16* __restrict__ dst, int n4) {
  int i = blockIdx.x * blockDim.x + threadIdx.x;
  if (i < n4) {
    float4 v = ((const float4*)src)[i];
    v4h h = { (_Float16)v.x, (_Float16)v.y, (_Float16)v.z, (_Float16)v.w };
    ((v4h*)dst)[i] = h;
  }
}

// ---- QKV GEMM: [8192,1024] @ [1024,3072] -> scatter to Q/K/V [B,H,N,D] ----
// Block tile 128(M) x 128(N), 256 threads = 8 waves, wave tile 32x64.
#define LDA 40   // 32 + 8 halves pad: 16B aligned rows, conflict-free
__global__ __launch_bounds__(256) void qkv_gemm_kernel(
    const _Float16* __restrict__ xh, const _Float16* __restrict__ wh,
    _Float16* __restrict__ Qh, _Float16* __restrict__ Kh,
    _Float16* __restrict__ Vh) {
  __shared__ _Float16 As[128 * LDA];
  __shared__ _Float16 Bs[128 * LDA];
  const int t = threadIdx.x;
  const int w = t >> 5, lane = t & 31;
  const int ln = lane & 15, hb = lane >> 4;
  const int m0 = blockIdx.x * 128, n0 = blockIdx.y * 128;
  const int wm = w & 3, wn = w >> 2;          // 4x2 wave grid, wave = 32x64
  v8f zero = {};
  v8f acc[2][4];
#pragma unroll
  for (int i = 0; i < 2; ++i)
#pragma unroll
    for (int j = 0; j < 4; ++j) acc[i][j] = zero;

  for (int k0 = 0; k0 < DIMC; k0 += 32) {
    __syncthreads();
#if HAS_TDM
    // A tile 128x32 via Tensor Data Mover; pad 32->40 halves per row
    // (interval 16 DW = code 3, amount 4 DW = code 3).
    if (w == 0)
      tdm_load_2d(lds_off_u32(As), &xh[(size_t)m0 * DIMC + k0],
                  32, 128, DIMC, 3, 3);
#else
#pragma unroll
    for (int u = 0; u < 2; ++u) {
      int c = t + u * 256;
      int row = c >> 2, cc = (c & 3) * 8;
      *(v8h*)&As[row * LDA + cc] =
          *(const v8h*)&xh[(size_t)(m0 + row) * DIMC + k0 + cc];
    }
#endif
    // B tile: 32(K)x128(N) -> transposed into Bs[n][k]
#pragma unroll
    for (int u = 0; u < 2; ++u) {
      int c = t + u * 256;
      int k = c >> 4, nc = (c & 15) * 8;
      v8h v = *(const v8h*)&wh[(size_t)(k0 + k) * 3072 + n0 + nc];
#pragma unroll
      for (int j = 0; j < 8; ++j) Bs[(nc + j) * LDA + k] = v[j];
    }
#if HAS_TDM
    __builtin_amdgcn_s_wait_tensorcnt(0);   // all waves: safe vs EXEC-masking
#endif
    __syncthreads();

    v16h a0 = ld_a_frag(&As[(wm * 32 + ln) * LDA + hb * 8]);
    v16h a1 = ld_a_frag(&As[(wm * 32 + 16 + ln) * LDA + hb * 8]);
    v16h b[4];
#pragma unroll
    for (int j = 0; j < 4; ++j)
      b[j] = ld_b_frag(&Bs[(wn * 64 + j * 16 + ln) * LDA + hb * 8]);
#pragma unroll
    for (int j = 0; j < 4; ++j) {
      acc[0][j] = mma16(a0, b[j], acc[0][j]);
      acc[1][j] = mma16(a1, b[j], acc[1][j]);
    }
  }

  // Epilogue: scatter into Q/K/V with [B,H,N,D] layout.
#pragma unroll
  for (int i = 0; i < 2; ++i)
#pragma unroll
    for (int j = 0; j < 4; ++j)
#pragma unroll
      for (int r = 0; r < 8; ++r) {
        int R  = m0 + wm * 32 + i * 16 + r + 8 * hb;   // global row in [0,8192)
        int cg = n0 + wn * 64 + j * 16 + ln;           // global col in [0,3072)
        int t3 = cg >> 10;                             // 0=q 1=k 2=v
        int h  = (cg >> 6) & 15;
        int d  = cg & 63;
        int b_ = R >> 11;
        int nr = R & 2047;
        _Float16* dst = (t3 == 0) ? Qh : ((t3 == 1) ? Kh : Vh);
        dst[(((size_t)b_ * NHEAD + h) * SEQN + nr) * HDIM + d] =
            (_Float16)acc[i][j][r];
      }
}

// ---- Flash attention: per (b,h), 64-query block, 4 waves x 16 rows --------
#define LDK 72   // 64 + 8 halves pad
__global__ __launch_bounds__(128) void flash_attn_kernel(
    const _Float16* __restrict__ Q, const _Float16* __restrict__ K,
    const _Float16* __restrict__ V, _Float16* __restrict__ O) {
  __shared__ _Float16 Ks[64 * LDK];      // K tile row-major (== B layout for QK^T)
  __shared__ _Float16 Vt[64 * LDK];      // V tile transposed: Vt[d][key]
  __shared__ _Float16 Ps[4 * 16 * LDK];  // per-wave P strip (D-layout -> A-layout)
  const int t = threadIdx.x, w = t >> 5, lane = t & 31;
  const int ln = lane & 15, hb = lane >> 4;
  const int bh = blockIdx.y;
  const int q0 = blockIdx.x * 64;
  const _Float16* Qb = Q + (size_t)bh * SEQN * HDIM;
  const _Float16* Kb = K + (size_t)bh * SEQN * HDIM;
  const _Float16* Vb = V + (size_t)bh * SEQN * HDIM;

  // Q fragments live in registers for the whole kernel; pre-scale by 1/sqrt(D).
  v16h qf0, qf1;
  {
    const _Float16* qp = Qb + (size_t)(q0 + w * 16 + ln) * HDIM + hb * 8;
    qf0 = ld_a_frag(qp);
    qf1 = ld_a_frag(qp + 32);
    const _Float16 sc = (_Float16)QKSCALE;
#pragma unroll
    for (int e = 0; e < 16; ++e) { qf0[e] *= sc; qf1[e] *= sc; }
  }

  float mrun[8], lrun[8];
  v8f zero = {};
  v8f o[4];
#pragma unroll
  for (int r = 0; r < 8; ++r) { mrun[r] = -3.0e38f; lrun[r] = 0.0f; }
#pragma unroll
  for (int d = 0; d < 4; ++d) o[d] = zero;

  for (int kt = 0; kt < SEQN / 64; ++kt) {
    __syncthreads();  // previous iteration's LDS reads done
#if HAS_TDM
    // K tile 64x64 via TDM; pad 64->72 halves per row
    // (interval 32 DW = code 4, amount 4 DW = code 3).
    if (w == 0)
      tdm_load_2d(lds_off_u32(Ks), Kb + (size_t)kt * 64 * HDIM,
                  64, 64, HDIM, 4, 3);
#endif
#pragma unroll
    for (int u = 0; u < 4; ++u) {
      int c = t + u * 128;
      int row = c >> 3, cc = (c & 7) * 8;
#if !HAS_TDM
      *(v8h*)&Ks[row * LDK + cc] =
          *(const v8h*)&Kb[(size_t)(kt * 64 + row) * HDIM + cc];
#endif
      v8h vv = *(const v8h*)&Vb[(size_t)(kt * 64 + row) * HDIM + cc];
#pragma unroll
      for (int j = 0; j < 8; ++j) Vt[(cc + j) * LDK + row] = vv[j];
    }
#if HAS_TDM
    __builtin_amdgcn_s_wait_tensorcnt(0);
#endif
    __syncthreads();

    // S = (Q*scale) @ K^T : 4 key sub-tiles, K-dim = D = 64 (2 wmma each)
    v8f s[4];
#pragma unroll
    for (int nt = 0; nt < 4; ++nt) {
      v8f c = zero;
      c = mma16(qf0, ld_b_frag(&Ks[(nt * 16 + ln) * LDK + hb * 8]), c);
      c = mma16(qf1, ld_b_frag(&Ks[(nt * 16 + ln) * LDK + hb * 8 + 32]), c);
      s[nt] = c;
    }

    // Online softmax: rows 0..7 live in lanes 0..15, rows 8..15 in lanes 16..31
#pragma unroll
    for (int r = 0; r < 8; ++r) {
      float mx = s[0][r];
#pragma unroll
      for (int nt = 1; nt < 4; ++nt) mx = fmaxf(mx, s[nt][r]);
      mx = fmaxf(mx, __shfl_xor(mx, 1));
      mx = fmaxf(mx, __shfl_xor(mx, 2));
      mx = fmaxf(mx, __shfl_xor(mx, 4));
      mx = fmaxf(mx, __shfl_xor(mx, 8));
      float mnew  = fmaxf(mrun[r], mx);
      float alpha = __expf(mrun[r] - mnew);
      mrun[r] = mnew;
      float rs = 0.0f;
#pragma unroll
      for (int nt = 0; nt < 4; ++nt) {
        float p = __expf(s[nt][r] - mnew);
        s[nt][r] = p;
        rs += p;
      }
      rs += __shfl_xor(rs, 1);
      rs += __shfl_xor(rs, 2);
      rs += __shfl_xor(rs, 4);
      rs += __shfl_xor(rs, 8);
      lrun[r] = lrun[r] * alpha + rs;
#pragma unroll
      for (int d = 0; d < 4; ++d) o[d][r] *= alpha;
    }

    // D-layout -> A-layout via per-wave LDS strip (no cross-wave sync needed)
    _Float16* Pw = &Ps[w * 16 * LDK];
#pragma unroll
    for (int nt = 0; nt < 4; ++nt)
#pragma unroll
      for (int r = 0; r < 8; ++r)
        Pw[(r + 8 * hb) * LDK + nt * 16 + ln] = (_Float16)s[nt][r];

    // O += P @ V  (K-dim = 64 keys, 2 chunks of 32)
#pragma unroll
    for (int c2 = 0; c2 < 2; ++c2) {
      v16h pf = ld_a_frag(&Pw[ln * LDK + hb * 8 + 32 * c2]);
#pragma unroll
      for (int d = 0; d < 4; ++d)
        o[d] = mma16(pf, ld_b_frag(&Vt[(d * 16 + ln) * LDK + hb * 8 + 32 * c2]),
                     o[d]);
    }
  }

  // Normalize and write O in [B, N, C] (C = h*64 + d) f16 for the proj GEMM.
  const int b = bh >> 4, h = bh & 15;
#pragma unroll
  for (int d = 0; d < 4; ++d)
#pragma unroll
    for (int r = 0; r < 8; ++r) {
      int qrow = q0 + w * 16 + r + 8 * hb;
      int col  = h * HDIM + d * 16 + ln;
      O[((size_t)b * SEQN + qrow) * DIMC + col] =
          (_Float16)(o[d][r] / lrun[r]);
    }
}

// ---- Proj GEMM: [8192,1024] @ [1024,1024] + bias -> f32 out ---------------
// Block tile 128x128, wave tile 32x64.
__global__ __launch_bounds__(256) void proj_gemm_kernel(
    const _Float16* __restrict__ Oh, const _Float16* __restrict__ wh,
    const float* __restrict__ bias, float* __restrict__ out) {
  __shared__ _Float16 As[128 * LDA];
  __shared__ _Float16 Bs[128 * LDA];
  const int t = threadIdx.x;
  const int w = t >> 5, lane = t & 31;
  const int ln = lane & 15, hb = lane >> 4;
  const int m0 = blockIdx.x * 128, n0 = blockIdx.y * 128;
  const int wm = w & 3, wn = w >> 2;
  v8f zero = {};
  v8f acc[2][4];
#pragma unroll
  for (int i = 0; i < 2; ++i)
#pragma unroll
    for (int j = 0; j < 4; ++j) acc[i][j] = zero;

  for (int k0 = 0; k0 < DIMC; k0 += 32) {
    __syncthreads();
#if HAS_TDM
    if (w == 0)
      tdm_load_2d(lds_off_u32(As), &Oh[(size_t)m0 * DIMC + k0],
                  32, 128, DIMC, 3, 3);
#else
#pragma unroll
    for (int u = 0; u < 2; ++u) {
      int c = t + u * 256;
      int row = c >> 2, cc = (c & 3) * 8;
      *(v8h*)&As[row * LDA + cc] =
          *(const v8h*)&Oh[(size_t)(m0 + row) * DIMC + k0 + cc];
    }
#endif
#pragma unroll
    for (int u = 0; u < 2; ++u) {
      int c = t + u * 256;
      int k = c >> 4, nc = (c & 15) * 8;
      v8h v = *(const v8h*)&wh[(size_t)(k0 + k) * DIMC + n0 + nc];
#pragma unroll
      for (int j = 0; j < 8; ++j) Bs[(nc + j) * LDA + k] = v[j];
    }
#if HAS_TDM
    __builtin_amdgcn_s_wait_tensorcnt(0);
#endif
    __syncthreads();

    v16h a0 = ld_a_frag(&As[(wm * 32 + ln) * LDA + hb * 8]);
    v16h a1 = ld_a_frag(&As[(wm * 32 + 16 + ln) * LDA + hb * 8]);
    v16h b[4];
#pragma unroll
    for (int j = 0; j < 4; ++j)
      b[j] = ld_b_frag(&Bs[(wn * 64 + j * 16 + ln) * LDA + hb * 8]);
#pragma unroll
    for (int j = 0; j < 4; ++j) {
      acc[0][j] = mma16(a0, b[j], acc[0][j]);
      acc[1][j] = mma16(a1, b[j], acc[1][j]);
    }
  }

#pragma unroll
  for (int i = 0; i < 2; ++i)
#pragma unroll
    for (int j = 0; j < 4; ++j) {
      int cg = n0 + wn * 64 + j * 16 + ln;
      float bv = bias[cg];
#pragma unroll
      for (int r = 0; r < 8; ++r) {
        int R = m0 + wm * 32 + i * 16 + r + 8 * hb;
        out[(size_t)R * DIMC + cg] = acc[i][j][r] + bv;
      }
    }
}

// ---------------------------------------------------------------------------
extern "C" void kernel_launch(void* const* d_in, const int* in_sizes, int n_in,
                              void* d_out, int out_size, void* d_ws,
                              size_t ws_size, hipStream_t stream) {
  (void)in_sizes; (void)n_in; (void)out_size; (void)ws_size;
  const float* x     = (const float*)d_in[0];
  const float* Wqkv  = (const float*)d_in[1];
  const float* Wproj = (const float*)d_in[2];
  const float* bproj = (const float*)d_in[3];
  float* out = (float*)d_out;
  char* ws = (char*)d_ws;

  // Workspace layout (f16), all offsets 256B-aligned. Total = 88 MiB.
  _Float16* xh     = (_Float16*)(ws + 0);          // 16 MiB: x as f16
  _Float16* Wqkvh  = (_Float16*)(ws + 16777216);   //  6 MiB
  _Float16* Wprojh = (_Float16*)(ws + 23068672);   //  2 MiB
  _Float16* Qh     = (_Float16*)(ws + 25165824);   // 16 MiB [B,H,N,D]
  _Float16* Kh     = (_Float16*)(ws + 41943040);   // 16 MiB
  _Float16* Vh     = (_Float16*)(ws + 58720256);   // 16 MiB
  _Float16* Oh     = (_Float16*)(ws + 75497472);   // 16 MiB [B,N,C]

  f32_to_f16_kernel<<<8192, 256, 0, stream>>>(x, xh, 2097152);
  f32_to_f16_kernel<<<3072, 256, 0, stream>>>(Wqkv, Wqkvh, 786432);
  f32_to_f16_kernel<<<1024, 256, 0, stream>>>(Wproj, Wprojh, 262144);

  qkv_gemm_kernel<<<dim3(64, 24), 256, 0, stream>>>(xh, Wqkvh, Qh, Kh, Vh);
  flash_attn_kernel<<<dim3(32, 64), 128, 0, stream>>>(Qh, Kh, Vh, Oh);
  proj_gemm_kernel<<<dim3(64, 8), 256, 0, stream>>>(Oh, Wprojh, bproj, out);
}
